// DisparityNetworkStage0_70119636074674
// MI455X (gfx1250) — matrix operand order repaired and checked by reading, not compiled, not run on hardware
//
#include <hip/hip_runtime.h>
#include <hip/hip_bf16.h>
#include <stdint.h>

// ---------------------------------------------------------------------------
// Problem constants (from the reference)
// ---------------------------------------------------------------------------
#define B_   8
#define H_   64
#define W_   120
#define C_   512
#define D_   12      // LOCAL_MAX_DISP
#define OH_  512
#define OW_  960

typedef __attribute__((ext_vector_type(16))) _Float16 v16h;
typedef __attribute__((ext_vector_type(8)))  float    v8f;

// ===========================================================================
// Kernel 1: cost volume  cost[b,i,j,k] = sum_c |fl[b,i,j,c] - fr[b,i,j-k,c]|
//           (fr treated as 0 for j < k).  Output packed f16.
// One workgroup per (b,i) row. Channel-chunked LDS staging so each fr element
// is read once from HBM and reused across all 12 disparities.
// ===========================================================================
#define CCHUNK 64   // channels per LDS chunk: 2 * 120*64*4B = 61.4 KB LDS

__global__ __launch_bounds__(256)
void cost_volume_kernel(const float* __restrict__ fl,
                        const float* __restrict__ fr,
                        _Float16* __restrict__ cost)
{
    __shared__ float sL[W_ * CCHUNK];
    __shared__ float sR[W_ * CCHUNK];

    const int b = blockIdx.x / H_;
    const int i = blockIdx.x % H_;
    const float* flrow = fl + (size_t)(b * H_ + i) * W_ * C_;
    const float* frrow = fr + (size_t)(b * H_ + i) * W_ * C_;

    // Hoist the (j, j-k) decomposition of each owned output out of the
    // channel loop (6 outputs per thread over the 1440 (j,k) pairs).
    int jv[6], jrv[6];
    #pragma unroll
    for (int s = 0; s < 6; ++s) {
        int o = threadIdx.x + s * 256;
        int oo = (o < W_ * D_) ? o : 0;
        jv[s]  = oo / D_;
        jrv[s] = jv[s] - (oo % D_);
    }

    float acc[6] = {0.f, 0.f, 0.f, 0.f, 0.f, 0.f};

    for (int c0 = 0; c0 < C_; c0 += CCHUNK) {
        // Prefetch next chunk (one cacheline per row per buffer).
        if (c0 + CCHUNK < C_) {
            int t = threadIdx.x;
            if (t < W_) {
                __builtin_prefetch(flrow + (size_t)t * C_ + c0 + CCHUNK, 0, 1);
            } else if (t < 2 * W_) {
                __builtin_prefetch(frrow + (size_t)(t - W_) * C_ + c0 + CCHUNK, 0, 1);
            }
        }
        // Cooperative load of this channel chunk for all 120 columns.
        const int nvec = W_ * CCHUNK / 4;  // float4 count
        for (int t = threadIdx.x; t < nvec; t += 256) {
            int j  = (t * 4) / CCHUNK;
            int cc = (t * 4) % CCHUNK;
            *(float4*)&sL[t * 4] = *(const float4*)&flrow[(size_t)j * C_ + c0 + cc];
            *(float4*)&sR[t * 4] = *(const float4*)&frrow[(size_t)j * C_ + c0 + cc];
        }
        __syncthreads();

        #pragma unroll
        for (int s = 0; s < 6; ++s) {
            if (threadIdx.x + s * 256 < W_ * D_) {
                float sum = acc[s];
                const float4* Lp = (const float4*)&sL[jv[s] * CCHUNK];
                if (jrv[s] >= 0) {
                    const float4* Rp = (const float4*)&sR[jrv[s] * CCHUNK];
                    #pragma unroll 4
                    for (int cc = 0; cc < CCHUNK / 4; ++cc) {
                        float4 l = Lp[cc], r = Rp[cc];
                        sum += fabsf(l.x - r.x) + fabsf(l.y - r.y)
                             + fabsf(l.z - r.z) + fabsf(l.w - r.w);
                    }
                } else {
                    #pragma unroll 4
                    for (int cc = 0; cc < CCHUNK / 4; ++cc) {
                        float4 l = Lp[cc];
                        sum += fabsf(l.x) + fabsf(l.y) + fabsf(l.z) + fabsf(l.w);
                    }
                }
                acc[s] = sum;
            }
        }
        __syncthreads();
    }

    _Float16* crow = cost + (size_t)(b * H_ + i) * W_ * D_;
    #pragma unroll
    for (int s = 0; s < 6; ++s) {
        int o = threadIdx.x + s * 256;
        if (o < W_ * D_) crow[o] = (_Float16)acc[s];
    }
}

// ===========================================================================
// Kernel 2 (template): 3x3x3 conv over (h, w, D) volume via WMMA im2col GEMM.
//   M = 16 output points (flattened (j,k) within a (b,i) row)
//   K = 27*CIN taps (padded to NCH*32), N = 16 (COUT valid columns)
//   Accumulated in f32 via V_WMMA_F32_16X16X32_F16.
//
// LDS slab is HALO-PADDED: [3 rows][W+2][D+2][CIN] with zeroed borders, plus
// one zeroed 8-byte slot for the K-padding region.  All im2col gathers are
// branch-free (v_cndmask address select to the zero slot, never an exec-mask
// branch).  For CIN=4 each afrag VGPR *pair* (v, v+1) covers all 4 cin values
// of one tap -> contiguous, 8B-aligned in LDS -> whole A fragment is just
// 4x ds_load_b64 per K-chunk.
//
// Fragment packing (ISA 7.12.2, wave32):
//   A 16x32 f16 : M = lane&15 ; Klocal = v<4 ? 2v+j+8*half : 16+2(v-4)+j+8*half
//   B 32x16 f16 : N = lane&15 ; Klocal = v<4 ? 2v+j+8*half : 16+2(v-4)+j+8*half
//   C 16x16 f32 : N = lane&15 ; M = r + 8*half
// ===========================================================================
template <int CIN, int COUT, bool RELU, bool OUT_F32>
__global__ __launch_bounds__(256)
void conv3d_wmma_kernel(const _Float16* __restrict__ in,
                        const float* __restrict__ wgt,   // (3,3,3,CIN,COUT) f32
                        const float* __restrict__ bias,  // (COUT)
                        void* __restrict__ outv)
{
    constexpr int K_VALID = 27 * CIN;
    constexpr int NCH     = (K_VALID + 31) / 32;       // K-chunks of 32
    constexpr int DP      = D_ + 2;                    // halo-padded disparity
    constexpr int RS      = (W_ + 2) * DP * CIN;       // halfs per padded slab row
    constexpr int ZIDX    = 3 * RS;                    // zeroed pad slot (8B-aligned)
    constexpr int ROWEL   = W_ * D_ * CIN;             // interior halfs per row

    __shared__ _Float16 sIn[3 * RS + 4];

    const int b = blockIdx.x / H_;
    const int i = blockIdx.x % H_;

    // --- zero-fill the whole slab (covers halos + K-pad slot) -------------
    {
        uint32_t* z32 = (uint32_t*)sIn;
        for (int t = threadIdx.x; t < (3 * RS + 4) / 2; t += 256) z32[t] = 0u;
    }
    __syncthreads();

    // --- stage interior of rows i-1, i, i+1 -------------------------------
    for (int r = 0; r < 3; ++r) {
        int ii = i - 1 + r;
        if (ii >= 0 && ii < H_) {
            const _Float16* src = in + (size_t)(b * H_ + ii) * ROWEL;
            if constexpr (CIN == 4) {
                // One aligned uint2 per (j,k) cell (4 contiguous halfs).
                const uint2* s8 = (const uint2*)src;
                uint2* d8 = (uint2*)sIn;
                for (int t = threadIdx.x; t < W_ * D_; t += 256) {
                    int j = t / D_;
                    int k = t - j * D_;
                    d8[(r * RS + ((1 + j) * DP + (1 + k)) * CIN) >> 2] = s8[t];
                }
            } else {
                for (int t = threadIdx.x; t < ROWEL; t += 256) {
                    int j = t / D_;
                    int k = t - j * D_;
                    sIn[r * RS + (1 + j) * DP + (1 + k)] = src[t];
                }
            }
        }
    }
    __syncthreads();

    const int lane = threadIdx.x & 31;
    const int wv   = threadIdx.x >> 5;
    const int half = lane >> 4;
    const int nlo  = lane & 15;

    // --- Build B (weight) fragments once per wave -------------------------
    v16h bfrag[NCH];
    #pragma unroll
    for (int c = 0; c < NCH; ++c) {
        #pragma unroll
        for (int v = 0; v < 8; ++v) {
            #pragma unroll
            for (int jj = 0; jj < 2; ++jj) {
                int kg = c * 32 + ((v < 4) ? (2 * v + jj + 8 * half) : (16 + 2 * (v - 4) + jj + 8 * half));   // global K index
                float wval = (kg < K_VALID && nlo < COUT)
                                 ? wgt[kg * COUT + nlo] : 0.f;
                bfrag[c][2 * v + jj] = (_Float16)wval;
            }
        }
    }
    const float bval = (nlo < COUT) ? bias[nlo] : 0.f;

    constexpr int PTS = W_ * D_;   // 1440 points per row
    constexpr int NT  = PTS / 16;  // 90 tiles of M=16

    const size_t rowbase = (size_t)(b * H_ + i) * PTS;

    for (int t = wv; t < NT; t += 8) {
        // Per-tile A-row position (independent of K-chunk): M = nlo.
        const int q    = t * 16 + nlo;
        const int j    = q / D_;
        const int k    = q - j * D_;
        const int cell = ((1 + j) * DP + (1 + k)) * CIN;   // halo-shifted base

        v8f acc = {};
        #pragma unroll
        for (int c = 0; c < NCH; ++c) {
            union { v16h h; uint2 d[4]; } af;
            if constexpr (CIN == 4) {
                // 4x ds_load_b64: each b64 = all 4 cin halfs of one tap.
                const uint2* s64 = (const uint2*)sIn;
                #pragma unroll
                for (int v2 = 0; v2 < 4; ++v2) {
                    int kl0 = (v2 < 2) ? (4 * v2 + 8 * half)
                                       : (16 + 4 * (v2 - 2) + 8 * half);
                    int kg0 = c * 32 + kl0;           // multiple of 4
                    int tap = kg0 >> 2;
                    int dh = tap / 9;
                    int rm = tap - 9 * dh;
                    int dw = rm / 3;
                    int dd = rm - 3 * dw;
                    int hidx = dh * RS + cell + (dw - 1) * (DP * CIN)
                             + (dd - 1) * CIN;        // cin0 == 0
                    hidx = (tap < 27) ? hidx : ZIDX;  // branch-free K-pad
                    af.d[v2] = s64[hidx >> 2];
                }
            } else {
                #pragma unroll
                for (int v = 0; v < 8; ++v) {
                    #pragma unroll
                    for (int jj = 0; jj < 2; ++jj) {
                        int kl = (v < 4) ? (2 * v + jj + 8 * half)
                                         : (16 + 2 * (v - 4) + jj + 8 * half);
                        int kg  = c * 32 + kl;        // CIN==1: tap == kg
                        int dh = kg / 9;
                        int rm = kg - 9 * dh;
                        int dw = rm / 3;
                        int dd = rm - 3 * dw;
                        int hidx = dh * RS + cell + (dw - 1) * DP + (dd - 1);
                        hidx = (kg < K_VALID) ? hidx : ZIDX;
                        af.h[2 * v + jj] = sIn[hidx];
                    }
                }
            }
            acc = __builtin_amdgcn_wmma_f32_16x16x32_f16(
                      false, af.h, false, bfrag[c], (short)0, acc, false, false);
        }

        // --- write back: output index is LINEAR in q (q = j*D + k) --------
        if (nlo < COUT) {
            const int q0 = t * 16 + 8 * half;
            #pragma unroll
            for (int r = 0; r < 8; ++r) {
                float o = acc[r] + bval;
                if (RELU) o = fmaxf(o, 0.f);
                size_t oidx = (rowbase + (size_t)(q0 + r)) * COUT + nlo;
                if constexpr (OUT_F32) ((float*)outv)[oidx] = o;
                else                   ((_Float16*)outv)[oidx] = (_Float16)o;
            }
        }
    }
}

// ===========================================================================
// Kernel 3: softmax(-cost) over D, expected disparity, * (HEIGHT*WIDTH)
// ===========================================================================
__global__ __launch_bounds__(256)
void softmax_exp_kernel(const float* __restrict__ cost2,
                        float* __restrict__ pred)
{
    int idx = blockIdx.x * 256 + threadIdx.x;
    if (idx >= B_ * H_ * W_) return;
    const float* c = cost2 + (size_t)idx * D_;
    float m = -c[0];
    #pragma unroll
    for (int k = 1; k < D_; ++k) m = fmaxf(m, -c[k]);
    float s = 0.f, num = 0.f;
    #pragma unroll
    for (int k = 0; k < D_; ++k) {
        float e = expf(-c[k] - m);
        s   += e;
        num += e * (float)k;
    }
    pred[idx] = (num / s) * ((float)OH_ * (float)OW_);
}

// ===========================================================================
// Kernel 4: bilinear resize (64,120) -> (512,960), half-pixel centers,
// edge-clamped (matches jax.image.resize 'bilinear').
// ===========================================================================
__global__ __launch_bounds__(256)
void resize_bilinear_kernel(const float* __restrict__ pred,
                            float* __restrict__ out)
{
    int idx = blockIdx.x * 256 + threadIdx.x;
    if (idx >= B_ * OH_ * OW_) return;
    int x = idx % OW_;
    int y = (idx / OW_) % OH_;
    int b = idx / (OW_ * OH_);

    float sy = (y + 0.5f) * ((float)H_ / (float)OH_) - 0.5f;
    float sx = (x + 0.5f) * ((float)W_ / (float)OW_) - 0.5f;
    float y0f = floorf(sy), x0f = floorf(sx);
    float fy = sy - y0f, fx = sx - x0f;
    int y0 = (int)y0f, x0 = (int)x0f;
    int y0c = min(max(y0, 0), H_ - 1);
    int y1c = min(max(y0 + 1, 0), H_ - 1);
    int x0c = min(max(x0, 0), W_ - 1);
    int x1c = min(max(x0 + 1, 0), W_ - 1);

    const float* p = pred + (size_t)b * H_ * W_;
    float p00 = p[y0c * W_ + x0c];
    float p01 = p[y0c * W_ + x1c];
    float p10 = p[y1c * W_ + x0c];
    float p11 = p[y1c * W_ + x1c];
    out[idx] = (1.f - fy) * ((1.f - fx) * p00 + fx * p01)
             +        fy  * ((1.f - fx) * p10 + fx * p11);
}

// ===========================================================================
// Host-side launch
// ===========================================================================
extern "C" void kernel_launch(void* const* d_in, const int* in_sizes, int n_in,
                              void* d_out, int out_size, void* d_ws, size_t ws_size,
                              hipStream_t stream)
{
    const float* feat_l = (const float*)d_in[0];
    const float* feat_r = (const float*)d_in[1];
    const float* w0 = (const float*)d_in[2];
    const float* b0 = (const float*)d_in[3];
    const float* w1 = (const float*)d_in[4];
    const float* b1 = (const float*)d_in[5];
    const float* w2 = (const float*)d_in[6];
    const float* b2 = (const float*)d_in[7];
    const float* w3 = (const float*)d_in[8];
    const float* b3 = (const float*)d_in[9];
    const float* wf = (const float*)d_in[10];
    const float* bf = (const float*)d_in[11];
    float* out = (float*)d_out;

    // Workspace layout (all offsets 256B-aligned), total ~15.7 MB
    char* ws = (char*)d_ws;
    _Float16* cost16 = (_Float16*)(ws);                             // 1,474,560 B
    _Float16* actA   = (_Float16*)(ws + 1474560);                   // 5,898,240 B
    _Float16* actB   = (_Float16*)(ws + 1474560 + 5898240);         // 5,898,240 B
    float*    cost2  = (float*)   (ws + 1474560 + 2 * 5898240);     // 2,949,120 B
    float*    pred   = (float*)   (ws + 1474560 + 2 * 5898240 + 2949120);
    (void)in_sizes; (void)n_in; (void)out_size; (void)ws_size;

    dim3 rowGrid(B_ * H_);   // 512 workgroups (one per (b,i) row)
    dim3 blk(256);

    // 1) cost volume (f16)
    cost_volume_kernel<<<rowGrid, blk, 0, stream>>>(feat_l, feat_r, cost16);

    // 2) conv stack via WMMA
    conv3d_wmma_kernel<1, 4, true,  false><<<rowGrid, blk, 0, stream>>>(cost16, w0, b0, actA);
    conv3d_wmma_kernel<4, 4, true,  false><<<rowGrid, blk, 0, stream>>>(actA,   w1, b1, actB);
    conv3d_wmma_kernel<4, 4, true,  false><<<rowGrid, blk, 0, stream>>>(actB,   w2, b2, actA);
    conv3d_wmma_kernel<4, 4, true,  false><<<rowGrid, blk, 0, stream>>>(actA,   w3, b3, actB);
    conv3d_wmma_kernel<4, 1, false, true ><<<rowGrid, blk, 0, stream>>>(actB,   wf, bf, cost2);

    // 3) softmax expectation
    int npred = B_ * H_ * W_;
    softmax_exp_kernel<<<dim3((npred + 255) / 256), blk, 0, stream>>>(cost2, pred);

    // 4) bilinear upsample to output
    int nout = B_ * OH_ * OW_;
    resize_bilinear_kernel<<<dim3((nout + 255) / 256), blk, 0, stream>>>(pred, out);
}
